// Alignment_89790586290734
// MI455X (gfx1250) — matrix operand order repaired and checked by reading, not compiled, not run on hardware
//
#include <hip/hip_runtime.h>
#include <hip/hip_bf16.h>
#include <math.h>

// Problem constants
#define Bsz 32
#define Lseq 1024
#define Dd 256

typedef __attribute__((ext_vector_type(16))) _Float16 v16h;
typedef __attribute__((ext_vector_type(8)))  float    v8f;
typedef __attribute__((ext_vector_type(4)))  unsigned int v4u;
typedef __attribute__((ext_vector_type(8)))  int     v8i;
typedef __attribute__((ext_vector_type(4)))  int     v4i;

union V16H { v16h v; _Float16 h[16]; uint4 q[2]; };
union V8F  { v8f  v; float f[8]; };

__device__ __forceinline__ v8f zero8() {
  v8f z = {0.f, 0.f, 0.f, 0.f, 0.f, 0.f, 0.f, 0.f};
  return z;
}

__device__ __forceinline__ v8f wmma16(v16h a, v16h b, v8f c) {
  // D = A(16x32 f16) * B(32x16 f16) + C(16x16 f32)
  return __builtin_amdgcn_wmma_f32_16x16x32_f16(
      /*neg_a=*/false, a, /*neg_b=*/false, b,
      /*c_mod=*/(short)0, c, /*reuse_a=*/false, /*reuse_b=*/false);
}

// A-fragment (16x32, f16) from a row-major f16 matrix (ld in halfs).
// ISA layout: lanes 0-15 -> M=lane, K in [0,8)+[16,24); lanes 16-31 -> M=lane-16,
// K in [8,16)+[24,32). Two contiguous 16B loads per lane.
__device__ __forceinline__ v16h load_a_f16(const _Float16* src, int ld,
                                           int row0, int k0, int lane) {
  int m  = lane & 15;
  int kb = (lane >> 4) << 3;           // 0 or 8
  const _Float16* p = src + (size_t)(row0 + m) * ld + k0 + kb;
  V16H r;
  r.q[0] = *(const uint4*)(p);         // K = kb .. kb+7
  r.q[1] = *(const uint4*)(p + 16);    // K = kb+16 .. kb+23
  return r.v;
}

// Same A-fragment but converting from f32 source on the fly.
__device__ __forceinline__ v16h load_a_f32cvt(const float* src, int ld,
                                              int row0, int k0, int lane) {
  int m  = lane & 15;
  int kb = (lane >> 4) << 3;
  const float* p = src + (size_t)(row0 + m) * ld + k0 + kb;
  V16H r;
#pragma unroll
  for (int i = 0; i < 8; ++i) {
    r.h[i]     = (_Float16)p[i];
    r.h[8 + i] = (_Float16)p[16 + i];
  }
  return r.v;
}

// B-fragment (32x16, f16) where column n of B is CONTIGUOUS in `src`
// (i.e. src holds B^T row-major with leading dim `ld`).
// ISA layout: lane -> N=lane&15; lanes 0-15 hold K in [0,8)+[16,24), lanes 16-31 K in [8,16)+[24,32).
__device__ __forceinline__ v16h load_b_cols(const _Float16* src, int ld,
                                            int n0, int k0, int lane) {
  int n  = lane & 15;
  int kh = (lane >> 4) << 3;           // 0 or 8
  const _Float16* p = src + (size_t)(n0 + n) * ld + k0 + kh;
  V16H r;
  r.q[0] = *(const uint4*)(p);
  r.q[1] = *(const uint4*)(p + 16);
  return r.v;
}

// ---------------------------------------------------------------------------
// TDM: issue one tensor_load_to_lds. Descriptors per ISA 8.3-8.5.
// All dims/strides in 8-byte units (data_size code 3). 2D tile (tile_d1 rows of
// tile_d0 units, row stride stride0 units), packed contiguously into LDS.
// For a 1-D copy pass tensor_d1=1, tile_d1=0.
// This toolchain exposes the 6-arg builtin:
//   (uint32x4 g0, int32x8 g1, int32x4 g2, int32x4 g3, int32x8 pad, i32 cpol)
// ---------------------------------------------------------------------------
__device__ __forceinline__ void tdm_load_to_lds(unsigned int lds_off, const void* gptr,
                                                unsigned int tensor_d0, unsigned int tensor_d1,
                                                unsigned int tile_d0, unsigned int tile_d1,
                                                unsigned int stride0) {
  unsigned long long ga = (unsigned long long)(uintptr_t)gptr;
  v4u g0;
  g0[0] = 1u;                                            // count=1, user descriptor
  g0[1] = lds_off;                                       // LDS byte address
  g0[2] = (unsigned int)(ga & 0xFFFFFFFFu);              // global_addr[31:0]
  g0[3] = (unsigned int)((ga >> 32) & 0x1FFFFFFu)        // global_addr[56:32]
        | (2u << 30);                                    // type = 2 ("image")
  v8i g1;
  g1[0] = (int)(3u << 16);                               // data_size = 8 bytes
  g1[1] = (int)((tensor_d0 & 0xFFFFu) << 16);            // tensor_dim0[15:0] @ [63:48]
  g1[2] = (int)(((tensor_d0 >> 16) & 0xFFFFu)            // tensor_dim0[31:16]
        | ((tensor_d1 & 0xFFFFu) << 16));                // tensor_dim1[15:0]
  g1[3] = (int)(((tensor_d1 >> 16) & 0xFFFFu)            // tensor_dim1[31:16]
        | ((tile_d0 & 0xFFFFu) << 16));                  // tile_dim0
  g1[4] = (int)(tile_d1 & 0xFFFFu);                      // tile_dim1 (tile_dim2=0)
  g1[5] = (int)stride0;                                  // tensor_dim0_stride[31:0]
  g1[6] = 0;                                             // stride0 hi / stride1 lo
  g1[7] = 0;
  v4i g2 = {0, 0, 0, 0};
  v4i g3 = {0, 0, 0, 0};
  v8i g4 = {0, 0, 0, 0, 0, 0, 0, 0};
  __builtin_amdgcn_tensor_load_to_lds(g0, g1, g2, g3, g4, 0);
}

// ---------------------------------------------------------------------------
// Kernel 1: F = tanh(X @ W), X:[B,L,D] f32, W:[D,H] f32, F:[B,L,H] f16
// grid = (L/128, B), block = 256 (8 waves; each wave owns a 16x256 tile)
// ---------------------------------------------------------------------------
__global__ void __launch_bounds__(256)
proj_tanh_kernel(const float* __restrict__ X, const float* __restrict__ W,
                 _Float16* __restrict__ F) {
  __shared__ __align__(16) _Float16 WT[256 * 32];   // W^T k-tile: [n][k], 16KB

  const int b    = blockIdx.y;
  const int tid  = threadIdx.x;
  const int wave = tid >> 5;
  const int lane = tid & 31;
  const int row0 = blockIdx.x * 128 + wave * 16;

  const float* Xb = X + (size_t)b * Lseq * Dd;
  _Float16*    Fb = F + (size_t)b * Lseq * Dd;

  v8f acc[16];
#pragma unroll
  for (int i = 0; i < 16; ++i) acc[i] = zero8();

  for (int kt = 0; kt < 8; ++kt) {
    const int k0 = kt * 32;
    __syncthreads();
    for (int idx = tid; idx < 32 * 256; idx += 256) {
      int k = idx >> 8, n = idx & 255;
      WT[n * 32 + k] = (_Float16)W[(size_t)(k0 + k) * Dd + n];
    }
    __syncthreads();

    v16h a = load_a_f32cvt(Xb, Dd, row0, k0, lane);
#pragma unroll
    for (int nf = 0; nf < 16; ++nf) {
      v16h bf = load_b_cols(WT, 32, nf * 16, 0, lane);
      acc[nf] = wmma16(a, bf, acc[nf]);
    }
  }

  const int h = lane >> 4, nlo = lane & 15;
#pragma unroll
  for (int nf = 0; nf < 16; ++nf) {
    V8F c; c.v = acc[nf];
#pragma unroll
    for (int r = 0; r < 8; ++r) {
      int row = row0 + r + 8 * h;
      Fb[(size_t)row * Dd + nf * 16 + nlo] = (_Float16)tanhf(c.f[r]);
    }
  }
}

// ---------------------------------------------------------------------------
// Kernel 2: transpose + convert. V:[B,L,D] f32 -> Vt:[B,D,L] f16.
// grid = (L/32, D/32, B), block = (32,8)
// ---------------------------------------------------------------------------
__global__ void __launch_bounds__(256)
transpose_cvt_kernel(const float* __restrict__ V, _Float16* __restrict__ Vt) {
  __shared__ float tile[32][33];
  const int b    = blockIdx.z;
  const int row0 = blockIdx.x * 32;   // L dim
  const int col0 = blockIdx.y * 32;   // D dim
  const float* Vb = V  + (size_t)b * Lseq * Dd;
  _Float16*   Vtb = Vt + (size_t)b * Dd * Lseq;

  for (int r = threadIdx.y; r < 32; r += 8)
    tile[r][threadIdx.x] = Vb[(size_t)(row0 + r) * Dd + col0 + threadIdx.x];
  __syncthreads();
  for (int c = threadIdx.y; c < 32; c += 8)
    Vtb[(size_t)(col0 + c) * Lseq + row0 + threadIdx.x] =
        (_Float16)tile[threadIdx.x][c];
}

// ---------------------------------------------------------------------------
// Kernel 3: fused flash-style alignment with TDM double-buffered staging.
//   Out[q,:] = softmax_k( FQ[q,:] . FK[k,:] ) @ V      (per batch)
// FQ/FK: [B,L,256] f16, Vt: [B,256,L] f16 (pre-transposed V), Out: [B,L,256] f32
// grid = (L/128, B), block = 256 (8 waves; wave owns 16 query rows x 256)
// ---------------------------------------------------------------------------
__global__ void __launch_bounds__(256)
flash_align_kernel(const _Float16* __restrict__ FQ,
                   const _Float16* __restrict__ FK,
                   const _Float16* __restrict__ Vt,
                   float* __restrict__ Out) {
  __shared__ __align__(16) _Float16 KsBuf[2][32 * 256];   // key tiles, 2x16KB
  __shared__ __align__(16) _Float16 VTBuf[2][256 * 32];   // V^T tiles, 2x16KB
  __shared__ __align__(16) _Float16 Ps[8][16 * 32];       // per-wave probs, 8KB

  const int b    = blockIdx.y;
  const int tid  = threadIdx.x;
  const int wave = tid >> 5;
  const int lane = tid & 31;
  const int row0 = blockIdx.x * 128 + wave * 16;

  const _Float16* FQb = FQ + (size_t)b * Lseq * Dd;
  const _Float16* FKb = FK + (size_t)b * Lseq * Dd;
  const _Float16* Vtb = Vt + (size_t)b * Dd * Lseq;

  // Persistent query fragments: 16 rows x 256 (8 K-tiles of 32)
  v16h qf[8];
#pragma unroll
  for (int kt = 0; kt < 8; ++kt) qf[kt] = load_a_f16(FQb, Dd, row0, kt * 32, lane);

  v8f O[16];
#pragma unroll
  for (int i = 0; i < 16; ++i) O[i] = zero8();
  float m[8], l[8];
#pragma unroll
  for (int r = 0; r < 8; ++r) { m[r] = -INFINITY; l[r] = 0.f; }

  // Prologue: DMA tile 0 into buffer 0 (wave 0 issues; TENSORcnt tracks).
  if (wave == 0) {
    // Keys: 32 rows x 256 halfs, contiguous 16KB -> 1-D copy of 2048 x 8B.
    tdm_load_to_lds((unsigned int)(uintptr_t)&KsBuf[0][0], FKb,
                    2048u, 1u, 2048u, 0u, 2048u);
    // V^T: 256 rows x 32 halfs (64B = 8 units), row stride 1024 halfs = 256 units.
    tdm_load_to_lds((unsigned int)(uintptr_t)&VTBuf[0][0], Vtb,
                    256u, 256u, 8u, 256u, 256u);
  }

  for (int q0 = 0; q0 < Lseq; q0 += 32) {
    const int buf = (q0 >> 5) & 1;
    if (wave == 0) __builtin_amdgcn_s_wait_tensorcnt(0);   // buffer `buf` ready
    __syncthreads();                                       // publish to all waves
    // Kick DMA for the next tile into the other buffer (overlaps with compute;
    // the barrier above guarantees everyone is done reading it).
    if (wave == 0 && q0 + 32 < Lseq) {
      tdm_load_to_lds((unsigned int)(uintptr_t)&KsBuf[buf ^ 1][0],
                      FKb + (size_t)(q0 + 32) * Dd, 2048u, 1u, 2048u, 0u, 2048u);
      tdm_load_to_lds((unsigned int)(uintptr_t)&VTBuf[buf ^ 1][0],
                      Vtb + (q0 + 32), 256u, 256u, 8u, 256u, 256u);
    }
    const _Float16* Kst = KsBuf[buf];
    const _Float16* VTt = VTBuf[buf];

    // S = FQ_tile (16x256) @ FK_tile^T (256x32) -> two 16x16 C-fragments
    v8f s0 = zero8(), s1 = zero8();
#pragma unroll
    for (int kt = 0; kt < 8; ++kt) {
      v16h b0 = load_b_cols(Kst, Dd, 0,  kt * 32, lane);
      v16h b1 = load_b_cols(Kst, Dd, 16, kt * 32, lane);
      s0 = wmma16(qf[kt], b0, s0);
      s1 = wmma16(qf[kt], b1, s1);
    }
    V8F S0; S0.v = s0;
    V8F S1; S1.v = s1;

    // Online softmax: C-fragment element r lives in row (r + 8*(lane>=16)),
    // N = lane&15 — reduce across each 16-lane half for per-row stats.
    float scale[8];
#pragma unroll
    for (int r = 0; r < 8; ++r) {
      float v = fmaxf(S0.f[r], S1.f[r]);
#pragma unroll
      for (int off = 8; off >= 1; off >>= 1) v = fmaxf(v, __shfl_xor(v, off, 32));
      float mn = fmaxf(m[r], v);
      scale[r] = __expf(m[r] - mn);
      S0.f[r] = __expf(S0.f[r] - mn);
      S1.f[r] = __expf(S1.f[r] - mn);
      float part = S0.f[r] + S1.f[r];
#pragma unroll
      for (int off = 8; off >= 1; off >>= 1) part += __shfl_xor(part, off, 32);
      l[r] = l[r] * scale[r] + part;
      m[r] = mn;
    }
#pragma unroll
    for (int nf = 0; nf < 16; ++nf) {
      V8F c; c.v = O[nf];
#pragma unroll
      for (int r = 0; r < 8; ++r) c.f[r] *= scale[r];
      O[nf] = c.v;
    }
    // C-layout -> A-layout via per-wave LDS scratch (wave-private, DS in-order)
    {
      _Float16* P = Ps[wave];
      const int h = lane >> 4, nlo = lane & 15;
#pragma unroll
      for (int r = 0; r < 8; ++r) {
        int row = r + 8 * h;
        P[row * 32 + nlo]      = (_Float16)S0.f[r];
        P[row * 32 + 16 + nlo] = (_Float16)S1.f[r];
      }
    }
    v16h ap = load_a_f16(Ps[wave], 32, 0, 0, lane);
    // O += P(16x32) @ V_tile(32x256)
#pragma unroll
    for (int nf = 0; nf < 16; ++nf) {
      v16h bv = load_b_cols(VTt, 32, nf * 16, 0, lane);
      O[nf] = wmma16(ap, bv, O[nf]);
    }
  }

  // Epilogue: Out = O / l
  float* Ob = Out + (size_t)b * Lseq * Dd;
  const int h = lane >> 4, nlo = lane & 15;
#pragma unroll
  for (int nf = 0; nf < 16; ++nf) {
    V8F c; c.v = O[nf];
#pragma unroll
    for (int r = 0; r < 8; ++r) {
      int row = row0 + r + 8 * h;
      Ob[(size_t)row * Dd + nf * 16 + nlo] = c.f[r] / l[r];
    }
  }
}

extern "C" void kernel_launch(void* const* d_in, const int* in_sizes, int n_in,
                              void* d_out, int out_size, void* d_ws, size_t ws_size,
                              hipStream_t stream) {
  const float* premises   = (const float*)d_in[0];   // [32,1024,256] f32
  const float* hypotheses = (const float*)d_in[1];   // [32,1024,256] f32
  const float* W_F        = (const float*)d_in[2];   // [256,256] f32

  const size_t elems = (size_t)Bsz * Lseq * Dd;      // 8,388,608
  float* betas  = (float*)d_out;
  float* alphas = betas + elems;

  _Float16* Fp = (_Float16*)d_ws;        // tanh(P@W)        16 MB
  _Float16* Fh = Fp + elems;             // tanh(H@W)        16 MB
  _Float16* Ht = Fh + elems;             // H^T f16 [B,D,L]  16 MB
  _Float16* Pt = Ht + elems;             // P^T f16 [B,D,L]  16 MB

  dim3 grid(Lseq / 128, Bsz), block(256);
  dim3 tgrid(Lseq / 32, Dd / 32, Bsz), tblock(32, 8);

  // Projections: F_p = tanh(P@W), F_h = tanh(H@W), stored f16.
  proj_tanh_kernel<<<grid, block, 0, stream>>>(premises,   W_F, Fp);
  proj_tanh_kernel<<<grid, block, 0, stream>>>(hypotheses, W_F, Fh);

  // Pre-transpose values to f16 [B,D,L] so the flash kernel can TDM-stage them.
  transpose_cvt_kernel<<<tgrid, tblock, 0, stream>>>(hypotheses, Ht);
  transpose_cvt_kernel<<<tgrid, tblock, 0, stream>>>(premises,   Pt);

  // betas : softmax over hypotheses (keys=F_h, values=H)
  flash_align_kernel<<<grid, block, 0, stream>>>(Fp, Fh, Ht, betas);
  // alphas: softmax over premises   (keys=F_p, values=P)
  flash_align_kernel<<<grid, block, 0, stream>>>(Fh, Fp, Pt, alphas);
}
